// Head_15710990368781
// MI455X (gfx1250) — compile-verified, matrix operand order checked by reading, not run on hardware
//
#include <hip/hip_runtime.h>
#include <hip/hip_bf16.h>

// Fused causal single-head attention for MI455X (gfx1250, wave32).
// One workgroup per batch element; q/k/v projections + flash-attention
// softmax fully fused, all intermediates in LDS, WMMA f16->f32 tensor ops.

typedef _Float16 v8h  __attribute__((ext_vector_type(8)));
typedef _Float16 v16h __attribute__((ext_vector_type(16)));
typedef float    v8f  __attribute__((ext_vector_type(8)));

#define B_  512
#define T_  256
#define C_  128
#define H_  16
#define NT  (T_ / 16)   // 16 row tiles of 16
#define NW  4           // waves per block (128 threads)

// Load a 16-bit WMMA A/B operand: two contiguous 8 x f16 (16B) LDS reads.
__device__ __forceinline__ v16h lds_ld_op(const _Float16* p0, const _Float16* p1) {
  v8h lo = *(const v8h*)p0;
  v8h hi = *(const v8h*)p1;
  return __builtin_shufflevector(lo, hi, 0,1,2,3,4,5,6,7,8,9,10,11,12,13,14,15);
}

// Row-wise reductions across the 16 lanes that hold one matrix row's N values.
// xor masks 1,2,4,8 stay inside each 16-lane half of the wave32.
__device__ __forceinline__ float redmax16(float v) {
  #pragma unroll
  for (int m = 1; m < 16; m <<= 1) v = fmaxf(v, __shfl_xor(v, m, 32));
  return v;
}
__device__ __forceinline__ float redsum16(float v) {
  #pragma unroll
  for (int m = 1; m < 16; m <<= 1) v += __shfl_xor(v, m, 32);
  return v;
}

__global__ __launch_bounds__(128, 1)
void attn_head_kernel(const float* __restrict__ x,
                      const float* __restrict__ Wq,
                      const float* __restrict__ Wk,
                      const float* __restrict__ Wv,
                      float* __restrict__ out) {
  // LDS staging (total ~56 KB of the 320 KB WGP LDS)
  __shared__ __align__(16) _Float16 sWT[3][H_ * C_];  // W^T, rows = h, contiguous over c
  __shared__ __align__(16) _Float16 sQ[T_ * 32];      // q * C^-0.5, [T][32], cols 16..31 = 0
  __shared__ __align__(16) _Float16 sK[T_ * 32];      // k,            [T][32], cols 16..31 = 0
  __shared__ __align__(16) _Float16 sVT[H_ * T_];     // v transposed, [H][T]
  __shared__ __align__(16) _Float16 sP[NW][16 * 32];  // per-wave softmax tile staging

  const int b    = blockIdx.x;
  const int tid  = threadIdx.x;
  const int lane = tid & 31;
  const int wave = tid >> 5;
  const int nidx = lane & 15;             // row (A) / column (B,D) index within tile
  const int hoff = (lane & 16) ? 8 : 0;   // lane-half K/M offset per WMMA layouts

  // ---- Stage W^T into LDS (f32 -> f16) and zero the K-padding of sQ/sK ----
  for (int idx = tid; idx < C_ * H_; idx += 128) {   // idx = c*16 + h
    const int c = idx >> 4, h = idx & 15;
    sWT[0][h * C_ + c] = (_Float16)Wq[idx];
    sWT[1][h * C_ + c] = (_Float16)Wk[idx];
    sWT[2][h * C_ + c] = (_Float16)Wv[idx];
  }
  {
    v8h z = {};
    for (int r = tid; r < T_; r += 128) {
      *(v8h*)&sQ[r * 32 + 16] = z; *(v8h*)&sQ[r * 32 + 24] = z;
      *(v8h*)&sK[r * 32 + 16] = z; *(v8h*)&sK[r * 32 + 24] = z;
    }
  }
  __syncthreads();

  // ---- Phase A: q,k,v projections (contraction over C=128, 4 K-chunks of 32) ----
  const float scale = 0.08838834764831845f;  // C^-0.5, folded into q
  for (int it = wave; it < NT; it += NW) {
    const int row = it * 16 + nidx;
    const float* xr = x + ((size_t)b * T_ + row) * C_;
    v16h a[4];
    #pragma unroll
    for (int c = 0; c < 4; ++c) {
      const int kb = c * 32;
      const float4 f0 = *(const float4*)(xr + kb + hoff);
      const float4 f1 = *(const float4*)(xr + kb + hoff + 4);
      const float4 f2 = *(const float4*)(xr + kb + 16 + hoff);
      const float4 f3 = *(const float4*)(xr + kb + 16 + hoff + 4);
      a[c][0]  = (_Float16)f0.x; a[c][1]  = (_Float16)f0.y;
      a[c][2]  = (_Float16)f0.z; a[c][3]  = (_Float16)f0.w;
      a[c][4]  = (_Float16)f1.x; a[c][5]  = (_Float16)f1.y;
      a[c][6]  = (_Float16)f1.z; a[c][7]  = (_Float16)f1.w;
      a[c][8]  = (_Float16)f2.x; a[c][9]  = (_Float16)f2.y;
      a[c][10] = (_Float16)f2.z; a[c][11] = (_Float16)f2.w;
      a[c][12] = (_Float16)f3.x; a[c][13] = (_Float16)f3.y;
      a[c][14] = (_Float16)f3.z; a[c][15] = (_Float16)f3.w;
    }
    v8f accq = {}, acck = {}, accv = {};
    #pragma unroll
    for (int c = 0; c < 4; ++c) {
      const int kb = c * 32;
      v16h bq = lds_ld_op(&sWT[0][nidx * C_ + kb + hoff], &sWT[0][nidx * C_ + kb + 16 + hoff]);
      v16h bk = lds_ld_op(&sWT[1][nidx * C_ + kb + hoff], &sWT[1][nidx * C_ + kb + 16 + hoff]);
      v16h bv = lds_ld_op(&sWT[2][nidx * C_ + kb + hoff], &sWT[2][nidx * C_ + kb + 16 + hoff]);
      accq = __builtin_amdgcn_wmma_f32_16x16x32_f16(false, a[c], false, bq, (short)0, accq, false, false);
      acck = __builtin_amdgcn_wmma_f32_16x16x32_f16(false, a[c], false, bk, (short)0, acck, false, false);
      accv = __builtin_amdgcn_wmma_f32_16x16x32_f16(false, a[c], false, bv, (short)0, accv, false, false);
    }
    // D layout: lane holds column nidx of rows {r+hoff}.
    #pragma unroll
    for (int r = 0; r < 8; ++r) {
      const int m = it * 16 + r + hoff;
      sQ[m * 32 + nidx]  = (_Float16)(accq[r] * scale);
      sK[m * 32 + nidx]  = (_Float16)acck[r];
      sVT[nidx * T_ + m] = (_Float16)accv[r];
    }
  }
  __syncthreads();

  // ---- Phase B: causal flash attention over j-tile pairs ----
  for (int it = wave; it < NT; it += NW) {
    v16h aq = lds_ld_op(&sQ[(it * 16 + nidx) * 32 + hoff],
                        &sQ[(it * 16 + nidx) * 32 + 16 + hoff]);
    float mrow[8], lrow[8];
    v8f o = {};
    #pragma unroll
    for (int r = 0; r < 8; ++r) { mrow[r] = -1e30f; lrow[r] = 0.0f; }

    for (int jj = 0; jj <= it; jj += 2) {   // wave-uniform loop: EXEC stays full
      v8f s0 = {}, s1 = {};
      {
        v16h bk = lds_ld_op(&sK[(jj * 16 + nidx) * 32 + hoff],
                            &sK[(jj * 16 + nidx) * 32 + 16 + hoff]);
        s0 = __builtin_amdgcn_wmma_f32_16x16x32_f16(false, aq, false, bk, (short)0, s0, false, false);
      }
      const bool has1 = (jj + 1) <= it;     // wave-uniform
      if (has1) {
        v16h bk = lds_ld_op(&sK[((jj + 1) * 16 + nidx) * 32 + hoff],
                            &sK[((jj + 1) * 16 + nidx) * 32 + 16 + hoff]);
        v8f z = {};
        s1 = __builtin_amdgcn_wmma_f32_16x16x32_f16(false, aq, false, bk, (short)0, z, false, false);
      } else {
        #pragma unroll
        for (int r = 0; r < 8; ++r) s1[r] = -1e30f;
      }
      // Causal mask on diagonal tiles: element (m = r+hoff, n = nidx) masked if n > m.
      if (jj == it) {
        #pragma unroll
        for (int r = 0; r < 8; ++r) if (nidx > r + hoff) s0[r] = -1e30f;
      }
      if (has1 && (jj + 1) == it) {
        #pragma unroll
        for (int r = 0; r < 8; ++r) if (nidx > r + hoff) s1[r] = -1e30f;
      }
      // Online softmax update (per-row stats replicated across each 16-lane half).
      #pragma unroll
      for (int r = 0; r < 8; ++r) {
        const float mt = redmax16(fmaxf(s0[r], s1[r]));
        const float mn = fmaxf(mrow[r], mt);
        const float alpha = __expf(mrow[r] - mn);
        const float p0 = __expf(s0[r] - mn);
        const float p1 = __expf(s1[r] - mn);
        lrow[r] = lrow[r] * alpha + redsum16(p0 + p1);
        o[r] *= alpha;
        mrow[r] = mn;
        sP[wave][(r + hoff) * 32 + nidx]      = (_Float16)p0;
        sP[wave][(r + hoff) * 32 + 16 + nidx] = (_Float16)p1;
      }
      // O += P(16x32) @ V(32x16); B operand from transposed V rows (contiguous).
      v16h ap = lds_ld_op(&sP[wave][nidx * 32 + hoff],
                          &sP[wave][nidx * 32 + 16 + hoff]);
      v16h bv = lds_ld_op(&sVT[nidx * T_ + jj * 16 + hoff],
                          &sVT[nidx * T_ + jj * 16 + 16 + hoff]);
      o = __builtin_amdgcn_wmma_f32_16x16x32_f16(false, ap, false, bv, (short)0, o, false, false);
    }
    // Epilogue: out[b, m, h] = o / l
    #pragma unroll
    for (int r = 0; r < 8; ++r) {
      const int m = it * 16 + r + hoff;
      out[((size_t)b * T_ + m) * H_ + nidx] = o[r] / lrow[r];
    }
  }
}

extern "C" void kernel_launch(void* const* d_in, const int* in_sizes, int n_in,
                              void* d_out, int out_size, void* d_ws, size_t ws_size,
                              hipStream_t stream) {
  (void)in_sizes; (void)n_in; (void)out_size; (void)d_ws; (void)ws_size;
  const float* x  = (const float*)d_in[0];
  const float* Wq = (const float*)d_in[1];
  const float* Wk = (const float*)d_in[2];
  const float* Wv = (const float*)d_in[3];
  float* out = (float*)d_out;
  attn_head_kernel<<<dim3(B_), dim3(128), 0, stream>>>(x, Wq, Wk, Wv, out);
}
